// Mamba2Block_39943195852851
// MI455X (gfx1250) — compile-verified, matrix operand order checked by reading, not run on hardware
//
#include <hip/hip_runtime.h>

#define D_MODEL 768
#define D_STATE 64
#define D_CONV  4
#define D_INNER 1536
#define SEQLEN  1024
#define NPROJ   (D_INNER + 2 * D_STATE)   // 1664

typedef __bf16 bf16;
typedef __attribute__((ext_vector_type(16))) __bf16 v16bf;
typedef __attribute__((ext_vector_type(8)))  float  v8f;
typedef unsigned int v4u __attribute__((ext_vector_type(4)));
typedef int v8i __attribute__((ext_vector_type(8)));
typedef int v4i __attribute__((ext_vector_type(4)));

#if __has_builtin(__builtin_amdgcn_tensor_load_to_lds)
#define USE_TDM 1
#if __has_include(<hip/amd_detail/amd_gfx1250_TDM.h>)
#define TDM_6ARG 1
#endif
#else
#define USE_TDM 0
#endif

// ---------------------------------------------------------------------------
// bf16 WMMA GEMM, TDM-staged + double-buffered.
//   C[M,N](f32) = alpha * A[M,K](bf16) @ (Bt[N,K](bf16))^T
// Block: 128 thr = 4 waves; block tile 64x128; wave tile 32x64 (2x4 WMMA);
// K-tile 64 (two K=32 steps -> 16 WMMA per wave per stage).
// Tiles are DMA'd into LDS by the Tensor Data Mover (row pad 128B+8B = 136B
// = LSTR bf16), double-buffered: tile i+1's TDM is issued while tile i
// computes; s_wait_tensorcnt 2 fences the current buffer only.
// reverseA: load mirrored row block ascending, flip LDS row at gather time.
// ---------------------------------------------------------------------------
#define BTM  64
#define BTN  128
#define KT   64
#define LSTR 68

#if USE_TDM
__device__ __forceinline__ void tdm_load_tile_bf16(
    unsigned lds_byte_off, const bf16* gsrc, unsigned rows,
    unsigned row_stride_elems, unsigned tens_d0, unsigned tens_d1)
{
  unsigned long ga = (unsigned long)gsrc;
  v4u g0;
  g0.x = 1u;                                    // count=1 (valid user D#)
  g0.y = lds_byte_off;                          // lds_addr
  g0.z = (unsigned)ga;                          // global_addr[31:0]
  g0.w = ((unsigned)(ga >> 32) & 0x1FFFFFFu) | (2u << 30);  // [56:32] | type=2
  v8i g1;
  // data_size=1 (2B), pad_enable, pad_interval code 4 (32 DWORDs = 128B row),
  // pad_amount code 1 (2 DWORDs = 8B)  ->  LDS row stride 136B = 68 bf16
  g1[0] = (int)((1u << 16) | (1u << 20) | (4u << 22) | (1u << 25));
  g1[1] = (int)((tens_d0 & 0xFFFFu) << 16);                       // dim0 lo16
  g1[2] = (int)((tens_d0 >> 16) | ((tens_d1 & 0xFFFFu) << 16));   // dim0 hi / dim1 lo
  g1[3] = (int)((tens_d1 >> 16) | ((unsigned)KT << 16));          // dim1 hi / tile_dim0
  g1[4] = (int)rows;                                              // tile_dim1 (tile_dim2=0)
  g1[5] = (int)row_stride_elems;                                  // dim0_stride lo32
  g1[6] = 0;
  g1[7] = 0;
  v4i z4 = {0, 0, 0, 0};
#ifdef TDM_6ARG
  v8i z8 = {0, 0, 0, 0, 0, 0, 0, 0};
  __builtin_amdgcn_tensor_load_to_lds(g0, g1, z4, z4, z8, 0);
#else
  __builtin_amdgcn_tensor_load_to_lds(g0, g1, z4, z4, 0);
#endif
}
#endif

__global__ __launch_bounds__(128) void gemm_bf16_wmma(
    const bf16* __restrict__ A,   // [M][K] bf16, lda = K
    const bf16* __restrict__ Bt,  // [N][K] bf16, ldb = K (pre-transposed)
    float* __restrict__ C,
    int M, int N, int K, int ldc,
    int reverseA, int flipRow, float alpha, int accumulate)
{
  __shared__ bf16 Ald[2][BTM * LSTR];   // [m][k], double buffered
  __shared__ bf16 Bld[2][BTN * LSTR];   // [n][k], double buffered

  const int tid  = threadIdx.x;
  const int lane = tid & 31;
  const int wave = tid >> 5;
  const int wm   = (wave >> 1) * 32;    // wave row offset (2 waves)
  const int wn   = (wave & 1)  * 64;    // wave col offset (2 waves)
  const int bm   = blockIdx.y * BTM;
  const int bn   = blockIdx.x * BTN;
  const int khalf = lane >> 4;
  const int l16   = lane & 15;

  const int srowA = reverseA ? (M - BTM - bm) : bm;
  const bf16* Abase = A  + (long)srowA * K;
  const bf16* Bbase = Bt + (long)bn    * K;

  v8f acc[2][4] = {};
  const int nk = K / KT;

#if USE_TDM
  if (wave == 0) {      // prologue: fill buffer 0
    tdm_load_tile_bf16((unsigned)(unsigned long)(void*)&Ald[0][0],
                       Abase, BTM, (unsigned)K, (unsigned)K, (unsigned)M);
    tdm_load_tile_bf16((unsigned)(unsigned long)(void*)&Bld[0][0],
                       Bbase, BTN, (unsigned)K, (unsigned)K, (unsigned)N);
  }
#endif

  for (int it = 0; it < nk; ++it) {
    const int cur = it & 1;
#if USE_TDM
    if (wave == 0) {
      if (it + 1 < nk) {   // issue next tile into the other buffer, then
        const int nxt = (it + 1) & 1;          // fence only the current one
        tdm_load_tile_bf16((unsigned)(unsigned long)(void*)&Ald[nxt][0],
                           Abase + (long)(it + 1) * KT, BTM,
                           (unsigned)K, (unsigned)K, (unsigned)M);
        tdm_load_tile_bf16((unsigned)(unsigned long)(void*)&Bld[nxt][0],
                           Bbase + (long)(it + 1) * KT, BTN,
                           (unsigned)K, (unsigned)K, (unsigned)N);
        __builtin_amdgcn_s_wait_tensorcnt(2);
      } else {
        __builtin_amdgcn_s_wait_tensorcnt(0);
      }
    }
#else
    {
      // conversion-free bf16 copy staging into buf[cur]
      const int r = tid >> 1, h = (tid & 1) * 32;     // A: 64 rows x 2 halves
      const uint2* ga = (const uint2*)(Abase + (long)r * K + it * KT + h);
      uint2* la = (uint2*)&Ald[cur][r * LSTR + h];
      #pragma unroll
      for (int j = 0; j < 8; ++j) la[j] = ga[j];
      const uint2* gb = (const uint2*)(Bbase + (long)tid * K + it * KT); // B: 128 rows
      uint2* lb = (uint2*)&Bld[cur][tid * LSTR];
      #pragma unroll
      for (int j = 0; j < 16; ++j) lb[j] = gb[j];
    }
#endif
    __syncthreads();

    #pragma unroll
    for (int ks = 0; ks < KT; ks += 32) {
      v16bf af[2], bfr[4];
      #pragma unroll
      for (int tm = 0; tm < 2; ++tm) {
        int r = wm + tm * 16 + l16;
        if (reverseA) r = BTM - 1 - r;
        const bf16* src = &Ald[cur][r * LSTR + ks + khalf * 8];
        #pragma unroll
        for (int i = 0; i < 16; ++i)
          af[tm][i] = src[((i >> 3) << 4) + (i & 7)];
      }
      #pragma unroll
      for (int tn = 0; tn < 4; ++tn) {
        const bf16* src = &Bld[cur][(wn + tn * 16 + l16) * LSTR + ks + khalf * 8];
        #pragma unroll
        for (int i = 0; i < 16; ++i)
          bfr[tn][i] = src[((i >> 3) << 4) + (i & 7)];
      }
      #pragma unroll
      for (int tm = 0; tm < 2; ++tm)
        #pragma unroll
        for (int tn = 0; tn < 4; ++tn)
          acc[tm][tn] = __builtin_amdgcn_wmma_f32_16x16x32_bf16(
              false, af[tm], false, bfr[tn], (short)0, acc[tm][tn], false, false);
    }
    __syncthreads();
  }

  const int rhalf = (lane >> 4) * 8;
  #pragma unroll
  for (int tm = 0; tm < 2; ++tm) {
    #pragma unroll
    for (int tn = 0; tn < 4; ++tn) {
      int coln = bn + wn + tn * 16 + l16;
      #pragma unroll
      for (int r = 0; r < 8; ++r) {
        int rowm = bm + wm + tm * 16 + rhalf + r;
        int orow = flipRow ? (M - 1 - rowm) : rowm;
        float v = alpha * acc[tm][tn][r];
        if (accumulate) C[(long)orow * ldc + coln] += v;
        else            C[(long)orow * ldc + coln]  = v;
      }
    }
  }
}

// ---------------------------------------------------------------------------
// One-time converters (per launch): W[K][N] f32 -> W^T[N][K] bf16, x -> bf16.
// ---------------------------------------------------------------------------
__global__ __launch_bounds__(256) void transpose_bf16_kernel(
    const float* __restrict__ W, bf16* __restrict__ WT, int K, int N)
{
  __shared__ float tile[32][33];
  int n0 = blockIdx.x * 32, k0 = blockIdx.y * 32;
  int tx = threadIdx.x, ty = threadIdx.y;          // (32, 8)
  #pragma unroll
  for (int j = 0; j < 32; j += 8)
    tile[ty + j][tx] = W[(long)(k0 + ty + j) * N + n0 + tx];
  __syncthreads();
  #pragma unroll
  for (int j = 0; j < 32; j += 8)
    WT[(long)(n0 + ty + j) * K + k0 + tx] = (bf16)tile[tx][ty + j];
}

__global__ __launch_bounds__(256) void f32_to_bf16_kernel(
    const float* __restrict__ in, bf16* __restrict__ out, long n)
{
  long i = (long)blockIdx.x * blockDim.x + threadIdx.x;
  if (i < n) out[i] = (bf16)in[i];
}

// ---------------------------------------------------------------------------
// Depthwise causal conv (width 4) + SiLU -> f32 (scan) + bf16 (GEMM2).
// ---------------------------------------------------------------------------
__global__ __launch_bounds__(256) void conv_silu_kernel(
    const float* __restrict__ xz, const float* __restrict__ cw,
    const float* __restrict__ cb, float* __restrict__ xc,
    bf16* __restrict__ xcbf)
{
  int d = blockIdx.x * blockDim.x + threadIdx.x;
  int t = blockIdx.y;
  float acc = cb[d];
  #pragma unroll
  for (int k = 0; k < D_CONV; ++k) {
    int tt = t + k - (D_CONV - 1);
    float xv = (tt >= 0) ? xz[(long)tt * (2 * D_INNER) + d] : 0.f;
    acc = fmaf(cw[d * D_CONV + k], xv, acc);
  }
  float s = acc * (1.f / (1.f + __expf(-acc)));
  long i = (long)t * D_INNER + d;
  xc[i]   = s;
  xcbf[i] = (bf16)s;
}

// ---------------------------------------------------------------------------
// Selective scan.  One wave32 per channel; each lane owns 2 of the 64 states.
// ---------------------------------------------------------------------------
__global__ __launch_bounds__(256) void scan_kernel(
    const float* __restrict__ proj, const float* __restrict__ xc,
    const float* __restrict__ A_log, float* __restrict__ ys)
{
  const int lane = threadIdx.x & 31;
  const int wave = threadIdx.x >> 5;
  const int d    = blockIdx.x * 8 + wave;
  const float A  = -__expf(A_log[d]);
  const int  n2  = lane * 2;

  float h0 = 0.f, h1 = 0.f;
  for (int t = 0; t < SEQLEN; ++t) {
    const float* pr = proj + (long)t * NPROJ;
    float dr    = pr[d];
    float delta = (dr > 20.f) ? dr : log1pf(__expf(dr));   // softplus
    float xv    = xc[(long)t * D_INNER + d];
    float dA    = __expf(delta * A);
    float dBx   = delta * xv;
    float2 Bv   = *(const float2*)(pr + D_INNER + n2);
    float2 Cv   = *(const float2*)(pr + D_INNER + D_STATE + n2);
    h0 = fmaf(dA, h0, dBx * Bv.x);
    h1 = fmaf(dA, h1, dBx * Bv.y);
    float p = Cv.x * h0 + Cv.y * h1;
    #pragma unroll
    for (int off = 16; off > 0; off >>= 1)
      p += __shfl_xor(p, off, 32);
    if (lane == 0) ys[(long)t * D_INNER + d] = p;
  }
}

// ---------------------------------------------------------------------------
// y = (ys + D * xc) * silu(z), emitted directly as bf16 (GEMM3 operand).
// ---------------------------------------------------------------------------
__global__ __launch_bounds__(256) void gate_kernel(
    const float* __restrict__ ys, const float* __restrict__ xc,
    const float* __restrict__ Dp, const float* __restrict__ xz,
    bf16* __restrict__ ygbf)
{
  int d = blockIdx.x * blockDim.x + threadIdx.x;
  int t = blockIdx.y;
  float z  = xz[(long)t * (2 * D_INNER) + D_INNER + d];
  float sz = z * (1.f / (1.f + __expf(-z)));
  long i = (long)t * D_INNER + d;
  ygbf[i] = (bf16)((ys[i] + Dp[d] * xc[i]) * sz);
}

// ---------------------------------------------------------------------------
extern "C" void kernel_launch(void* const* d_in, const int* in_sizes, int n_in,
                              void* d_out, int out_size, void* d_ws, size_t ws_size,
                              hipStream_t stream) {
  (void)in_sizes; (void)n_in; (void)out_size; (void)ws_size;

  const float* x      = (const float*)d_in[0];
  const float* W_in   = (const float*)d_in[1];
  const float* conv_w = (const float*)d_in[2];
  const float* conv_b = (const float*)d_in[3];
  const float* W_x    = (const float*)d_in[4];
  const float* A_log  = (const float*)d_in[5];
  const float* Dp     = (const float*)d_in[6];
  const float* W_out  = (const float*)d_in[7];
  float*       out    = (float*)d_out;

  char* w = (char*)d_ws;
  float* xz   = (float*)w;  w += (long)SEQLEN * 2 * D_INNER * 4;   // 1024x3072
  float* xc   = (float*)w;  w += (long)SEQLEN * D_INNER * 4;       // 1024x1536
  float* proj = (float*)w;  w += (long)SEQLEN * NPROJ * 4;         // 1024x1664
  float* ys   = (float*)w;  w += (long)SEQLEN * D_INNER * 4;       // 1024x1536
  bf16* xbf   = (bf16*)w;   w += (long)SEQLEN * D_MODEL * 2;       // 1024x768
  bf16* xcbf  = (bf16*)w;   w += (long)SEQLEN * D_INNER * 2;       // 1024x1536
  bf16* ygbf  = (bf16*)w;   w += (long)SEQLEN * D_INNER * 2;       // 1024x1536
  bf16* WinT  = (bf16*)w;   w += (long)(2 * D_INNER) * D_MODEL * 2;// 3072x768
  bf16* WxT   = (bf16*)w;   w += (long)NPROJ * D_INNER * 2;        // 1664x1536
  bf16* WoutT = (bf16*)w;   w += (long)D_MODEL * D_INNER * 2;      // 768x1536

  // ---- one-time (per launch) operand preparation ----
  f32_to_bf16_kernel<<<(SEQLEN * D_MODEL + 255) / 256, 256, 0, stream>>>(
      x, xbf, (long)SEQLEN * D_MODEL);
  transpose_bf16_kernel<<<dim3(2 * D_INNER / 32, D_MODEL / 32), dim3(32, 8), 0, stream>>>(
      W_in, WinT, D_MODEL, 2 * D_INNER);
  transpose_bf16_kernel<<<dim3(NPROJ / 32, D_INNER / 32), dim3(32, 8), 0, stream>>>(
      W_x, WxT, D_INNER, NPROJ);
  transpose_bf16_kernel<<<dim3(D_MODEL / 32, D_INNER / 32), dim3(32, 8), 0, stream>>>(
      W_out, WoutT, D_INNER, D_MODEL);

  for (int dir = 0; dir < 2; ++dir) {
    // 1) in-projection: xz = (dir ? flip(x) : x) @ W_in
    gemm_bf16_wmma<<<dim3((2 * D_INNER) / BTN, SEQLEN / BTM), 128, 0, stream>>>(
        xbf, WinT, xz, SEQLEN, 2 * D_INNER, D_MODEL,
        2 * D_INNER, dir, 0, 1.f, 0);

    // 2) causal depthwise conv + SiLU
    conv_silu_kernel<<<dim3(D_INNER / 256, SEQLEN), 256, 0, stream>>>(
        xz, conv_w, conv_b, xc, xcbf);

    // 3) x-projection: proj = xc @ W_x -> [delta | B | C]
    gemm_bf16_wmma<<<dim3(NPROJ / BTN, SEQLEN / BTM), 128, 0, stream>>>(
        xcbf, WxT, proj, SEQLEN, NPROJ, D_INNER,
        NPROJ, 0, 0, 1.f, 0);

    // 4) selective scan
    scan_kernel<<<dim3(D_INNER / 8), 256, 0, stream>>>(proj, xc, A_log, ys);

    // 5) skip connection + SiLU(z) gating -> bf16
    gate_kernel<<<dim3(D_INNER / 256, SEQLEN), 256, 0, stream>>>(
        ys, xc, Dp, xz, ygbf);

    // 6) out-projection: out (f)  = 0.5 * yg @ W_out
    //                    out (b) += 0.5 * flip(yg @ W_out)
    gemm_bf16_wmma<<<dim3(D_MODEL / BTN, SEQLEN / BTM), 128, 0, stream>>>(
        ygbf, WoutT, out, SEQLEN, D_MODEL, D_INNER,
        D_MODEL, 0, dir, 0.5f, dir);
  }
}
